// CausalSelfAttention_63505386438963
// MI455X (gfx1250) — matrix operand order repaired and checked by reading, not compiled, not run on hardware
//
#include <hip/hip_runtime.h>

// CDNA5 (gfx1250) WMMA types
typedef __attribute__((ext_vector_type(16))) __bf16 v16bf;
typedef __attribute__((ext_vector_type(8)))  float  v8f;
typedef __attribute__((ext_vector_type(4)))  unsigned int v4u;
typedef __attribute__((ext_vector_type(8)))  int v8i;
typedef __attribute__((ext_vector_type(4)))  int v4i;

#define BATCH 2
#define SEQ   4096
#define CDIM  512
#define NQKV  1536
#define MTOT  (BATCH * SEQ)   // 8192

#if __has_builtin(__builtin_amdgcn_tensor_load_to_lds)
#define HAVE_TDM 1
#else
#define HAVE_TDM 0
#endif

// ---------------------------------------------------------------------------
// helpers
// ---------------------------------------------------------------------------
static __device__ __forceinline__ unsigned short f2bf(float f) {
  unsigned int u = __float_as_uint(f);
  u += 0x7fffu + ((u >> 16) & 1u);   // round-to-nearest-even
  return (unsigned short)(u >> 16);
}

union FragU { uint4 q[2]; v16bf v; };

// A-matrix 16x32 bf16 fragment (ISA 7.12.2): lane row = lane&15,
// K pairs {0..3, 8..11} + 4*(lane>=16). Each half = 4 contiguous dwords ->
// two ds_load_b128 (row base must be 16B aligned).
static __device__ __forceinline__ v16bf frag_a(const unsigned short* row, int lh) {
  FragU f;
  const uint4* p = (const uint4*)row;
  f.q[0] = p[lh];        // pairs lh*4 .. lh*4+3
  f.q[1] = p[lh + 2];    // pairs lh*4+8 .. lh*4+11
  return f.v;
}

// B-matrix 32x16 bf16 fragment: lane col = lane&15, same K order as frag_a
// -> pairs {lh*4..lh*4+3, lh*4+8..lh*4+11}: two ds_load_b128.
static __device__ __forceinline__ v16bf frag_b(const unsigned short* row, int lh) {
  FragU f;
  const uint4* p = (const uint4*)row;
  f.q[0] = p[lh];
  f.q[1] = p[lh + 2];
  return f.v;
}

static __device__ __forceinline__ v8f wmma_bf16(v16bf a, v16bf b, v8f c) {
  return __builtin_amdgcn_wmma_f32_16x16x32_bf16(false, a, false, b, (short)0, c,
                                                 false, false);
}

#if HAVE_TDM
// Issue a 2D TDM tile load (global -> LDS), D# per CDNA5 ISA ch.8.
// All dims/strides in data_size=8B units; pad fields per-spec (DWORD encoded).
// amdgpu-toolchain (clang-23) 6-arg form:
//   (v4u g0, v8i g1, v4i g2, v4i g3, v8i, i32 cpol)
static __device__ __forceinline__ void tdm_load_2d(
    unsigned lds_off, const void* gptr, unsigned tensor_d0, unsigned tensor_d1,
    unsigned tile_d0, unsigned tile_d1, unsigned stride0, unsigned pad_interval,
    unsigned pad_amount) {
  unsigned long long ga = (unsigned long long)gptr;
  v4u g0;
  g0[0] = 1u;                                         // count=1, user mode
  g0[1] = lds_off;                                    // lds_addr (bytes)
  g0[2] = (unsigned)ga;                               // global_addr[31:0]
  g0[3] = (unsigned)((ga >> 32) & 0x1FFFFFFu) | (2u << 30);  // addr[56:32]|type=2
  v8i g1;
  g1[0] = (int)((3u << 16) | (1u << 20) | (pad_interval << 22) | (pad_amount << 25));
  g1[1] = (int)(tensor_d0 << 16);                     // tensor_dim0[15:0]
  g1[2] = (int)((tensor_d0 >> 16) | (tensor_d1 << 16));
  g1[3] = (int)((tensor_d1 >> 16) | (tile_d0 << 16)); // tile_dim0
  g1[4] = (int)tile_d1;                               // tile_dim1 (tile_dim2=0)
  g1[5] = (int)stride0;                               // tensor_dim0_stride[31:0]
  g1[6] = 0;
  g1[7] = 0;
  v4i z4 = {0, 0, 0, 0};
  v8i z8 = {0, 0, 0, 0, 0, 0, 0, 0};
  __builtin_amdgcn_tensor_load_to_lds(g0, g1, z4, z4, z8, 0);
}
#endif

// ---------------------------------------------------------------------------
// Kernel 1: qkv = x @ w_qkv + b_qkv  -> q (scaled 1/sqrt(c), bf16),
//           k (bf16 row-major), v transposed [b][c][n] (bf16)
// 256 threads (8 waves), tile 64(M) x 128(N), staged K step 64.
// ---------------------------------------------------------------------------
#define STR_G 72   // GEMM LDS stride (144B rows, 16B aligned)

__global__ __launch_bounds__(256) void k_qkv_proj(
    const float* __restrict__ x, const float* __restrict__ w,
    const float* __restrict__ bias, unsigned short* __restrict__ qb,
    unsigned short* __restrict__ kb, unsigned short* __restrict__ vtb) {
  __shared__ __align__(16) unsigned short As[64 * STR_G];   // [m][k]
  __shared__ __align__(16) unsigned short Bs[128 * STR_G];  // [n][k] transposed

  const int tid = threadIdx.x;
  const int lane = tid & 31;
  const int wv = tid >> 5;
  const int l16 = lane & 15, lh = lane >> 4;
  const int wm = wv >> 2, wn = wv & 3;
  const int m0 = blockIdx.x * 64;
  const int n0 = blockIdx.y * 128;

  v8f acc[2][2] = {};

  for (int k0 = 0; k0 < CDIM; k0 += 64) {
    // stage A (64x64 f32 -> bf16)
#pragma unroll
    for (int i = 0; i < 4; ++i) {
      int task = tid + 256 * i;            // 1024 float4 tasks
      int r = task >> 4, kq = (task & 15) * 4;
      const float4 f = *(const float4*)(x + (size_t)(m0 + r) * CDIM + k0 + kq);
      unsigned short* d = &As[r * STR_G + kq];
      d[0] = f2bf(f.x); d[1] = f2bf(f.y); d[2] = f2bf(f.z); d[3] = f2bf(f.w);
    }
    // stage B (64x128 f32 -> bf16, transposed into [n][k])
#pragma unroll
    for (int i = 0; i < 8; ++i) {
      int task = tid + 256 * i;            // 2048 float4 tasks
      int kk = task >> 5, ng = (task & 31) * 4;
      const float4 f = *(const float4*)(w + (size_t)(k0 + kk) * NQKV + n0 + ng);
      Bs[(ng + 0) * STR_G + kk] = f2bf(f.x);
      Bs[(ng + 1) * STR_G + kk] = f2bf(f.y);
      Bs[(ng + 2) * STR_G + kk] = f2bf(f.z);
      Bs[(ng + 3) * STR_G + kk] = f2bf(f.w);
    }
    __syncthreads();
#pragma unroll
    for (int kc = 0; kc < 2; ++kc) {
      v16bf a0 = frag_a(&As[(wm * 32 + l16) * STR_G + kc * 32], lh);
      v16bf a1 = frag_a(&As[(wm * 32 + 16 + l16) * STR_G + kc * 32], lh);
      v16bf b0 = frag_b(&Bs[(wn * 32 + l16) * STR_G + kc * 32], lh);
      v16bf b1 = frag_b(&Bs[(wn * 32 + 16 + l16) * STR_G + kc * 32], lh);
      acc[0][0] = wmma_bf16(a0, b0, acc[0][0]);
      acc[0][1] = wmma_bf16(a0, b1, acc[0][1]);
      acc[1][0] = wmma_bf16(a1, b0, acc[1][0]);
      acc[1][1] = wmma_bf16(a1, b1, acc[1][1]);
    }
    __syncthreads();
  }

  // epilogue: region is block-uniform (n0 multiple of 128; regions 512 wide)
  const float qscale = 0.044194173824159216f;  // 1/sqrt(512)
  float bs[2];
  bs[0] = bias[n0 + wn * 32 + l16];
  bs[1] = bias[n0 + wn * 32 + 16 + l16];
  const int region = n0 >> 9;   // 0:q 1:k 2:v  (scalar branch)
  if (region == 0) {
#pragma unroll
    for (int i = 0; i < 2; ++i)
#pragma unroll
      for (int j = 0; j < 2; ++j)
#pragma unroll
        for (int v = 0; v < 8; ++v) {
          int m = m0 + wm * 32 + i * 16 + v + 8 * lh;
          int n = n0 + wn * 32 + j * 16 + l16;
          qb[(size_t)m * CDIM + n] = f2bf((acc[i][j][v] + bs[j]) * qscale);
        }
  } else if (region == 1) {
#pragma unroll
    for (int i = 0; i < 2; ++i)
#pragma unroll
      for (int j = 0; j < 2; ++j)
#pragma unroll
        for (int v = 0; v < 8; ++v) {
          int m = m0 + wm * 32 + i * 16 + v + 8 * lh;
          int n = n0 - CDIM + wn * 32 + j * 16 + l16;
          kb[(size_t)m * CDIM + n] = f2bf(acc[i][j][v] + bs[j]);
        }
  } else {
#pragma unroll
    for (int i = 0; i < 2; ++i)
#pragma unroll
      for (int j = 0; j < 2; ++j)
#pragma unroll
        for (int v = 0; v < 8; ++v) {
          int m = m0 + wm * 32 + i * 16 + v + 8 * lh;
          int ch = n0 - 2 * CDIM + wn * 32 + j * 16 + l16;
          int bb = m >> 12, nr = m & (SEQ - 1);
          vtb[((size_t)bb * CDIM + ch) * SEQ + nr] = f2bf(acc[i][j][v] + bs[j]);
        }
  }
}

// ---------------------------------------------------------------------------
// Kernel 2: flash attention. One block per 64 query rows, 8 waves.
// K/V^T/Q tiles staged by the Tensor Data Mover (TENSORcnt tracked); the V
// DMA overlaps the S-GEMM + softmax phase (wait tensorcnt<=1 before S,
// tensorcnt==0 only before PV).
// ---------------------------------------------------------------------------
#define STR_QK 520   // 512 + 8 bf16 pad  (1040B rows; TDM pad 7/3)
#define STR_VT 72    // 64 + 8 bf16 pad   (144B rows;  TDM pad 4/3)
#define STR_P  72
#define STR_S  68

#define QS_OFF   0u
#define KS_OFF   (QS_OFF + 64u * STR_QK * 2u)      // 66560
#define VTS_OFF  (KS_OFF + 64u * STR_QK * 2u)      // 133120
#define PS_OFF   (VTS_OFF + 512u * STR_VT * 2u)    // 206848
#define SS_OFF   (PS_OFF + 64u * STR_P * 2u)       // 216064
#define STAT_OFF (SS_OFF + 64u * STR_S * 4u)       // 233472
#define SMEM_BYTES (STAT_OFF + 3u * 64u * 4u)      // 234240

__global__ __launch_bounds__(256) void k_flash_attn(
    const unsigned short* __restrict__ qb, const unsigned short* __restrict__ kb,
    const unsigned short* __restrict__ vtb, unsigned short* __restrict__ yb,
    float* __restrict__ denom) {
  extern __shared__ __align__(16) unsigned char smem_raw[];
  unsigned short* Qs = (unsigned short*)(smem_raw + QS_OFF);
  unsigned short* Ks = (unsigned short*)(smem_raw + KS_OFF);
  unsigned short* Vts = (unsigned short*)(smem_raw + VTS_OFF);
  unsigned short* Ps = (unsigned short*)(smem_raw + PS_OFF);
  float* Ss = (float*)(smem_raw + SS_OFF);
  float* m_st = (float*)(smem_raw + STAT_OFF);
  float* l_st = m_st + 64;
  float* al_st = l_st + 64;

  const int tid = threadIdx.x;
  const int lane = tid & 31;
  const int wv = tid >> 5;                // owns y columns [wv*64, wv*64+64)
  const int l16 = lane & 15, lh = lane >> 4;
  const int mt = blockIdx.x & 63;
  const int bb = blockIdx.x >> 6;
  const int m0 = mt * 64;
  const size_t rowbase = (size_t)(bb * SEQ + m0);
  const int rg = wv >> 1, cpair = wv & 1;

  if (tid < 64) { m_st[tid] = -3.0e38f; l_st[tid] = 0.0f; }

#if HAVE_TDM
  if (wv == 0) {   // Q tile 64x512 bf16: rows 1024B, LDS rows 1040B
    tdm_load_2d(QS_OFF, qb + rowbase * CDIM, 128, 64, 128, 64, 128, 7, 3);
  }
#else
#pragma unroll
  for (int i = 0; i < 8; ++i) {
    int task = tid + 256 * i;
    int r = task >> 5, seg = (task & 31) * 16;
    const uint4* src = (const uint4*)(qb + (rowbase + r) * CDIM + seg);
    uint4* dst = (uint4*)&Qs[r * STR_QK + seg];
    dst[0] = src[0]; dst[1] = src[1];
  }
#endif

  v8f yacc[4][4] = {};

  for (int kt = 0; kt <= mt; ++kt) {
    const int k0g = kt * 64;
    const size_t krow = (size_t)(bb * SEQ + k0g);
#if HAVE_TDM
    if (wv == 0) {
      // K tile 64x512 bf16
      tdm_load_2d(KS_OFF, kb + krow * CDIM, 128, 64, 128, 64, 128, 7, 3);
      // V^T tile 512x64 bf16 out of [c][n]: rows 8192B, tile rows 128B
      tdm_load_2d(VTS_OFF, vtb + (size_t)bb * CDIM * SEQ + k0g,
                  1024, 512, 16, 512, 1024, 4, 3);
      __builtin_amdgcn_s_wait_tensorcnt(1);   // K (and Q on iter 0) landed
    }
#else
#pragma unroll
    for (int i = 0; i < 8; ++i) {
      int task = tid + 256 * i;
      int r = task >> 5, seg = (task & 31) * 16;
      const uint4* src = (const uint4*)(kb + (krow + r) * CDIM + seg);
      uint4* dst = (uint4*)&Ks[r * STR_QK + seg];
      dst[0] = src[0]; dst[1] = src[1];
    }
#pragma unroll
    for (int i = 0; i < 8; ++i) {
      int task = tid + 256 * i;
      int ch = task >> 2, seg = (task & 3) * 16;
      const uint4* src =
          (const uint4*)(vtb + ((size_t)bb * CDIM + ch) * SEQ + k0g + seg);
      uint4* dst = (uint4*)&Vts[ch * STR_VT + seg];
      dst[0] = src[0]; dst[1] = src[1];
    }
#endif
    __syncthreads();

    // S = Q * K^T : each wave two 16x16 tiles, K-loop over c=512
    v8f s0 = {}, s1 = {};
#pragma unroll 4
    for (int ks = 0; ks < 16; ++ks) {
      v16bf a = frag_a(&Qs[(rg * 16 + l16) * STR_QK + ks * 32], lh);
      v16bf b0 = frag_b(&Ks[((cpair * 2 + 0) * 16 + l16) * STR_QK + ks * 32], lh);
      v16bf b1 = frag_b(&Ks[((cpair * 2 + 1) * 16 + l16) * STR_QK + ks * 32], lh);
      s0 = wmma_bf16(a, b0, s0);
      s1 = wmma_bf16(a, b1, s1);
    }
    // causal mask + spill S (f32) to LDS
#pragma unroll
    for (int t = 0; t < 2; ++t) {
      int scol = (cpair * 2 + t) * 16 + l16;
      int gk = k0g + scol;
#pragma unroll
      for (int v = 0; v < 8; ++v) {
        int srow = rg * 16 + v + 8 * lh;
        float val = (t == 0) ? s0[v] : s1[v];
        if (gk > m0 + srow) val = -3.0e38f;
        Ss[srow * STR_S + scol] = val;
      }
    }
    __syncthreads();

    // online softmax: 4 threads per row (same wave), shfl_xor reductions
    {
      int r = tid >> 2, seg = (tid & 3) * 16;
      float vals[16];
      float mx = -3.0e38f;
#pragma unroll
      for (int j = 0; j < 16; ++j) {
        vals[j] = Ss[r * STR_S + seg + j];
        mx = fmaxf(mx, vals[j]);
      }
      mx = fmaxf(mx, __shfl_xor(mx, 1, 32));
      mx = fmaxf(mx, __shfl_xor(mx, 2, 32));
      float mold = m_st[r];
      float mnew = fmaxf(mold, mx);
      float sum = 0.0f;
#pragma unroll
      for (int j = 0; j < 16; ++j) {
        float p = __expf(vals[j] - mnew);
        sum += p;
        Ps[r * STR_P + seg + j] = f2bf(p);
      }
      sum += __shfl_xor(sum, 1, 32);
      sum += __shfl_xor(sum, 2, 32);
      if ((tid & 3) == 0) {
        al_st[r] = __expf(mold - mnew);
        m_st[r] = mnew;
        l_st[r] = l_st[r] * al_st[r] + sum;
      }
    }
#if HAVE_TDM
    if (wv == 0) __builtin_amdgcn_s_wait_tensorcnt(0);  // V landed
#endif
    __syncthreads();

    // rescale running Y by alpha, then Y += P * V
#pragma unroll
    for (int rt = 0; rt < 4; ++rt)
#pragma unroll
      for (int v = 0; v < 8; ++v) {
        float al = al_st[rt * 16 + v + 8 * lh];
#pragma unroll
        for (int bt = 0; bt < 4; ++bt) yacc[rt][bt][v] *= al;
      }
#pragma unroll
    for (int ks = 0; ks < 2; ++ks) {
      v16bf bfr[4];
#pragma unroll
      for (int bt = 0; bt < 4; ++bt)
        bfr[bt] = frag_b(&Vts[(wv * 64 + bt * 16 + l16) * STR_VT + ks * 32], lh);
#pragma unroll
      for (int rt = 0; rt < 4; ++rt) {
        v16bf a = frag_a(&Ps[(rt * 16 + l16) * STR_P + ks * 32], lh);
#pragma unroll
        for (int bt = 0; bt < 4; ++bt)
          yacc[rt][bt] = wmma_bf16(a, bfr[bt], yacc[rt][bt]);
      }
    }
    __syncthreads();
  }

  // finalize: y /= denom, store bf16; emit softmax denominator
#pragma unroll
  for (int rt = 0; rt < 4; ++rt)
#pragma unroll
    for (int v = 0; v < 8; ++v) {
      int row = rt * 16 + v + 8 * lh;
      float inv = 1.0f / l_st[row];
#pragma unroll
      for (int bt = 0; bt < 4; ++bt) {
        int col = wv * 64 + bt * 16 + l16;
        yb[(rowbase + row) * CDIM + col] = f2bf(yacc[rt][bt][v] * inv);
      }
    }
  if (tid < 64) denom[bb * SEQ + m0 + tid] = l_st[tid];
}

// ---------------------------------------------------------------------------
// Kernel 3: out = y @ w_o + b_o  (y bf16, out f32)
// ---------------------------------------------------------------------------
__global__ __launch_bounds__(256) void k_out_proj(
    const unsigned short* __restrict__ yb, const float* __restrict__ w,
    const float* __restrict__ bias, float* __restrict__ out) {
  __shared__ __align__(16) unsigned short As[64 * STR_G];
  __shared__ __align__(16) unsigned short Bs[128 * STR_G];

  const int tid = threadIdx.x;
  const int lane = tid & 31;
  const int wv = tid >> 5;
  const int l16 = lane & 15, lh = lane >> 4;
  const int wm = wv >> 2, wn = wv & 3;
  const int m0 = blockIdx.x * 64;
  const int n0 = blockIdx.y * 128;

  v8f acc[2][2] = {};

  for (int k0 = 0; k0 < CDIM; k0 += 64) {
    // stage A (64x64 bf16): b128 load -> b128 LDS store
#pragma unroll
    for (int i = 0; i < 2; ++i) {
      int task = tid + 256 * i;            // 512 tasks x 8 bf16
      int r = task >> 3, kq = (task & 7) * 8;
      const uint4 f = *(const uint4*)(yb + (size_t)(m0 + r) * CDIM + k0 + kq);
      *(uint4*)&As[r * STR_G + kq] = f;    // 144B rows: 16B aligned
    }
    // stage B (64x128 f32 -> bf16, transposed)
#pragma unroll
    for (int i = 0; i < 8; ++i) {
      int task = tid + 256 * i;
      int kk = task >> 5, ng = (task & 31) * 4;
      const float4 f = *(const float4*)(w + (size_t)(k0 + kk) * CDIM + n0 + ng);
      Bs[(ng + 0) * STR_G + kk] = f2bf(f.x);
      Bs[(ng + 1) * STR_G + kk] = f2bf(f.y);
      Bs[(ng + 2) * STR_G + kk] = f2bf(f.z);
      Bs[(ng + 3) * STR_G + kk] = f2bf(f.w);
    }
    __syncthreads();
#pragma unroll
    for (int kc = 0; kc < 2; ++kc) {
      v16bf a0 = frag_a(&As[(wm * 32 + l16) * STR_G + kc * 32], lh);
      v16bf a1 = frag_a(&As[(wm * 32 + 16 + l16) * STR_G + kc * 32], lh);
      v16bf b0 = frag_b(&Bs[(wn * 32 + l16) * STR_G + kc * 32], lh);
      v16bf b1 = frag_b(&Bs[(wn * 32 + 16 + l16) * STR_G + kc * 32], lh);
      acc[0][0] = wmma_bf16(a0, b0, acc[0][0]);
      acc[0][1] = wmma_bf16(a0, b1, acc[0][1]);
      acc[1][0] = wmma_bf16(a1, b0, acc[1][0]);
      acc[1][1] = wmma_bf16(a1, b1, acc[1][1]);
    }
    __syncthreads();
  }

#pragma unroll
  for (int i = 0; i < 2; ++i)
#pragma unroll
    for (int j = 0; j < 2; ++j)
#pragma unroll
      for (int v = 0; v < 8; ++v) {
        int m = m0 + wm * 32 + i * 16 + v + 8 * lh;
        int n = n0 + wn * 32 + j * 16 + l16;
        out[(size_t)m * CDIM + n] = acc[i][j][v] + bias[n];
      }
}

// ---------------------------------------------------------------------------
// launch
// ---------------------------------------------------------------------------
extern "C" void kernel_launch(void* const* d_in, const int* in_sizes, int n_in,
                              void* d_out, int out_size, void* d_ws,
                              size_t ws_size, hipStream_t stream) {
  const float* x = (const float*)d_in[0];
  const float* w_qkv = (const float*)d_in[1];
  const float* b_qkv = (const float*)d_in[2];
  const float* w_o = (const float*)d_in[3];
  const float* b_o = (const float*)d_in[4];

  float* out = (float*)d_out;                    // [b,n,c] f32
  float* denom = out + (size_t)MTOT * CDIM;      // [b,n] f32

  // workspace: q | k | v^T | y, all bf16 [8192 x 512] => 32 MB total
  unsigned short* qb = (unsigned short*)d_ws;
  unsigned short* kb = qb + (size_t)MTOT * CDIM;
  unsigned short* vtb = kb + (size_t)MTOT * CDIM;
  unsigned short* yb = vtb + (size_t)MTOT * CDIM;

  k_qkv_proj<<<dim3(MTOT / 64, NQKV / 128), 256, 0, stream>>>(x, w_qkv, b_qkv,
                                                              qb, kb, vtb);

  k_flash_attn<<<dim3(MTOT / 64), 256, SMEM_BYTES, stream>>>(qb, kb, vtb, yb,
                                                             denom);

  k_out_proj<<<dim3(MTOT / 64, CDIM / 128), 256, 0, stream>>>(yb, w_o, b_o, out);
}
